// GraphAttentionPool_9328668966995
// MI455X (gfx1250) — compile-verified, matrix operand order checked by reading, not run on hardware
//
#include <hip/hip_runtime.h>
#include <hip/hip_bf16.h>

// ---------------------------------------------------------------------------
// GraphAttentionPool for MI455X (gfx1250, wave32, WMMA)
//   N=500000 nodes, D=256 features, H=128 hidden, B=1024 graphs
// Kernel 1: logits = tanh(x@W1+b1)@W2+b2  via v_wmma_f32_16x16x32_bf16
// Kernel 2: per-segment softmax + gated pooling (batch is sorted)
// ---------------------------------------------------------------------------

#define DD 256   // feature dim (K of GEMM)
#define HH 128   // hidden dim  (N of GEMM)

typedef __attribute__((ext_vector_type(16))) __bf16 v16bf;
typedef __attribute__((ext_vector_type(8)))  float  v8f;

union BF16Frag {
    v16bf v;
    unsigned short u[16];
    uint4 q[2];
};

// ---------------------------------------------------------------------------
// Kernel 1: per-node logits. One wave (32 lanes) owns 16 rows of x.
// W1 staged transposed in LDS as bf16: sW1t[n*DD + k], 64 KB.
// ---------------------------------------------------------------------------
__global__ __launch_bounds__(256) void gap_logits_kernel(
    const float* __restrict__ x,     // (N, DD)
    const float* __restrict__ W1,    // (DD, HH) row-major
    const float* __restrict__ b1,    // (HH)
    const float* __restrict__ W2,    // (HH)
    const float* __restrict__ b2,    // (1)
    float* __restrict__ logits,      // (N)  [gate region of d_out, reused]
    int N)
{
    __shared__ __bf16 sW1t[DD * HH];   // transposed bf16 W1: [n][k]

    const int tid = threadIdx.x;

    // Cooperative fp32 -> bf16 convert + transpose of W1 into LDS.
    for (int i = tid; i < DD * HH; i += 256) {
        int n = i >> 8;          // 0..127
        int k = i & 255;         // 0..255
        sW1t[i] = (__bf16)W1[k * HH + n];   // sW1t[n*DD + k]
    }
    __syncthreads();

    const int wave  = tid >> 5;
    const int lane  = tid & 31;
    const int mrow  = lane & 15;    // row / column index within 16-wide tile
    const int khalf = lane >> 4;    // 0: K 0..15 of chunk, 1: K 16..31

    const long row0 = ((long)blockIdx.x * 8 + wave) * 16;
    if (row0 >= N) return;

    const float bias2 = b2[0];

    v8f acc[8];
#pragma unroll
    for (int t = 0; t < 8; ++t) acc[t] = (v8f){0,0,0,0,0,0,0,0};

    const float* xrow = x + (row0 + mrow) * DD;

    for (int kb = 0; kb < DD; kb += 32) {
        // ---- A fragment: rows row0..row0+15, K = kb+khalf*16 .. +15 (bf16)
        BF16Frag a;
        const float4* xp4 = (const float4*)(xrow + kb + khalf * 16);
#pragma unroll
        for (int j = 0; j < 4; ++j) {
            float4 f = xp4[j];
            a.v[4*j+0] = (__bf16)f.x;   // native v_cvt_*_bf16_f32 path
            a.v[4*j+1] = (__bf16)f.y;
            a.v[4*j+2] = (__bf16)f.z;
            a.v[4*j+3] = (__bf16)f.w;
        }
        // hint the next K-tile of the x stream into cache
        if (kb + 32 < DD) __builtin_prefetch(xrow + kb + 32, 0, 1);

        // ---- 8 N-tiles of W1, each a 32x16 bf16 B fragment from LDS
#pragma unroll
        for (int t = 0; t < 8; ++t) {
            BF16Frag bfr;
            const uint4* wp = (const uint4*)
                (&sW1t[(t * 16 + mrow) * DD + kb + khalf * 16]);
            bfr.q[0] = wp[0];
            bfr.q[1] = wp[1];
            acc[t] = __builtin_amdgcn_wmma_f32_16x16x32_bf16(
                false, a.v, false, bfr.v, (short)0, acc[t], false, false);
        }
    }

    // ---- fused bias + tanh + dot(h, W2), reduced across each 16-lane half.
    // C layout: lanes 0-15 hold M=r (VGPR r), lanes 16-31 hold M=r+8; N=lane&15.
    float p[8];
#pragma unroll
    for (int r = 0; r < 8; ++r) p[r] = 0.0f;

#pragma unroll
    for (int t = 0; t < 8; ++t) {
        const int n  = t * 16 + mrow;
        const float b1v = b1[n];
        const float w2v = W2[n];
#pragma unroll
        for (int r = 0; r < 8; ++r)
            p[r] += tanhf(acc[t][r] + b1v) * w2v;
    }

#pragma unroll
    for (int m = 1; m < 16; m <<= 1) {
#pragma unroll
        for (int r = 0; r < 8; ++r)
            p[r] += __shfl_xor(p[r], m, 32);
    }

    if (mrow == 0) {
        const long base = row0 + 8 * khalf;   // lane0 -> rows 0..7, lane16 -> 8..15
#pragma unroll
        for (int r = 0; r < 8; ++r)
            logits[base + r] = p[r] + bias2;
    }
}

// ---------------------------------------------------------------------------
// Kernel 2: one block per graph. Segment softmax + gated pooling.
// logits_gate holds logits on entry, gate on exit (in-place).
// ---------------------------------------------------------------------------
__global__ __launch_bounds__(256) void gap_pool_kernel(
    const float* __restrict__ x,        // (N, DD)
    const int*   __restrict__ batch,    // (N) sorted
    float* __restrict__ logits_gate,    // (N) in: logits, out: gate
    float* __restrict__ pooled,         // (B, DD)
    int N)
{
    const int b   = blockIdx.x;
    const int tid = threadIdx.x;

    __shared__ float red[256];
    __shared__ float sg[256];

    // lower_bound over sorted batch ids
    auto lb = [&](int v) {
        int lo = 0, hi = N;
        while (lo < hi) {
            int mid = (lo + hi) >> 1;
            if (batch[mid] < v) lo = mid + 1; else hi = mid;
        }
        return lo;
    };
    const int start = lb(b);
    const int end   = lb(b + 1);

    // ---- pass A: segment max of logits
    float mx = -3.402823466e38f;
    for (int i = start + tid; i < end; i += 256)
        mx = fmaxf(mx, logits_gate[i]);
    red[tid] = mx;
    __syncthreads();
    for (int s = 128; s > 0; s >>= 1) {
        if (tid < s) red[tid] = fmaxf(red[tid], red[tid + s]);
        __syncthreads();
    }
    mx = red[0];
    __syncthreads();

    // ---- pass B: sum of exp(logit - max)
    float sm = 0.0f;
    for (int i = start + tid; i < end; i += 256)
        sm += __expf(logits_gate[i] - mx);
    red[tid] = sm;
    __syncthreads();
    for (int s = 128; s > 0; s >>= 1) {
        if (tid < s) red[tid] += red[tid + s];
        __syncthreads();
    }
    const float inv = 1.0f / red[0];
    __syncthreads();

    // ---- pass C: write gate (in place) + accumulate pooled feature sums.
    // Thread d = tid owns feature column d (DD == blockDim.x == 256).
    float acc = 0.0f;
    for (int base = start; base < end; base += 256) {
        const int cnt = min(256, end - base);
        if (tid < cnt) {
            float g = __expf(logits_gate[base + tid] - mx) * inv;
            sg[tid] = g;
            logits_gate[base + tid] = g;   // gate output overwrites logit
        }
        __syncthreads();
        for (int j = 0; j < cnt; ++j)
            acc += x[(long)(base + j) * DD + tid] * sg[j];
        __syncthreads();
    }
    pooled[(long)b * DD + tid] = acc;     // zero for empty segments
}

// ---------------------------------------------------------------------------
extern "C" void kernel_launch(void* const* d_in, const int* in_sizes, int n_in,
                              void* d_out, int out_size, void* d_ws, size_t ws_size,
                              hipStream_t stream) {
    const float* x     = (const float*)d_in[0];   // (N, 256)
    const float* W1    = (const float*)d_in[1];   // (256, 128)
    const float* b1    = (const float*)d_in[2];   // (128)
    const float* W2    = (const float*)d_in[3];   // (128, 1)
    const float* b2    = (const float*)d_in[4];   // (1)
    const int*   batch = (const int*)d_in[5];     // (N)

    const int N = in_sizes[5];
    const int B = (out_size - N) / DD;            // d_out = pooled(B,DD) ++ gate(N)

    float* pooled = (float*)d_out;
    float* gate   = (float*)d_out + (size_t)B * DD;  // logits scratch -> gate

    const int blocks1 = (N + 127) / 128;          // 8 waves/block, 16 rows/wave
    gap_logits_kernel<<<blocks1, 256, 0, stream>>>(x, W1, b1, W2, b2, gate, N);
    gap_pool_kernel<<<B, 256, 0, stream>>>(x, batch, gate, pooled, N);
}
